// KimiK25TextAttention_55662776156366
// MI455X (gfx1250) — matrix operand order repaired and checked by reading, not compiled, not run on hardware
//
#include <hip/hip_runtime.h>
#include <math.h>

// ---------------- problem constants ----------------
#define S_LEN 2048
#define H_DIM 4096
#define NH_   32
#define QLR_  1536
#define KVLR_ 512
#define DN_   128
#define DR_   64
#define DV_   128
#define QD_   192
#define NQD_  (NH_ * QD_)          // 6144
#define NKV_  (NH_ * (DN_ + DV_))  // 8192
#define CKVW_ (KVLR_ + DR_)        // 576

typedef __attribute__((ext_vector_type(16))) __bf16 v16bf;
typedef __attribute__((ext_vector_type(8)))  float  v8f;
typedef __attribute__((ext_vector_type(4))) unsigned int u32x4;
typedef __attribute__((ext_vector_type(8))) int         i32x8;
typedef __attribute__((ext_vector_type(4))) int         i32x4;

// LDS byte offset of a __shared__ object (generic -> AS3 -> int)
#define LDS_OFF(p) \
  ((unsigned)(unsigned long long)(__attribute__((address_space(3))) const void*)(const void*)(p))

// ---------------------------------------------------------------------------
// WMMA fragment helpers (CDNA5 wave32 layouts, CDNA5 ISA §7.12.2)
//  A (16x32 bf16): lane L holds row m=L%16; VGPR pair p holds
//       k0 = (p<4 ? 2p : 16+2(p-4)) + 8*(L/16)  and k0+1
//  B (32x16 bf16): lane L holds col n=L%16; element i holds
//       k = (i<8 ? i : 8+i) + 8*(L/16)
// ---------------------------------------------------------------------------
__device__ __forceinline__ v16bf frag_a_from_row(const __bf16* row, int half) {
  v16bf r;
#pragma unroll
  for (int p = 0; p < 8; ++p) {
    const int k0 = ((p < 4) ? (2 * p) : (16 + 2 * (p - 4))) + half * 8;
    r[2 * p]     = row[k0];
    r[2 * p + 1] = row[k0 + 1];
  }
  return r;
}

__device__ __forceinline__ v16bf frag_b_from_row(const __bf16* row, int half) {
  v16bf r;
  const __bf16* p = row + half * 8;
#pragma unroll
  for (int i = 0; i < 16; ++i) r[i] = p[(i < 8) ? i : (8 + i)];
  return r;
}

__device__ __forceinline__ v8f wmma_bf16(v16bf a, v16bf b, v8f c) {
  return __builtin_amdgcn_wmma_f32_16x16x32_bf16(false, a, false, b,
                                                 (short)0, c, false, false);
}

// ---------------------------------------------------------------------------
// Tensor Data Mover descriptors (CDNA5 ISA)
// group0: [1:0]=count=1, [63:32]=lds_addr, [120:64]=global_addr, [127:126]=2
// group1: [17:16]=data_size(1->2B), [20]=pad_en, [24:22]=pad_interval,
//         [31:25]=pad_amount, [79:48]=tensor_dim0, [111:80]=tensor_dim1,
//         [127:112]=tile_dim0, [143:128]=tile_dim1, [207:160]=dim0_stride
// Padding: 1 DWORD inserted per 16 DWORDs stored (pad_interval=3, amount=0)
//   -> bf16 element (row,d) of a W-elem row lands at byte
//      row*(2W + W/8) + 2d + 4*(d>>5)
// ---------------------------------------------------------------------------
__device__ __forceinline__ u32x4 tdm_g0(unsigned lds_off, const void* gaddr) {
  const unsigned long long ga = (unsigned long long)gaddr;
  u32x4 g;
  g[0] = 1u;  // count=1, user descriptor
  g[1] = lds_off;
  g[2] = (unsigned)(ga & 0xffffffffu);
  g[3] = (unsigned)((ga >> 32) & 0x01ffffffu) | (2u << 30);  // type=2 (image)
  return g;
}

__device__ __forceinline__ i32x8 tdm_g1_2d(unsigned tdim0, unsigned tdim1,
                                           unsigned tile0, unsigned tile1,
                                           unsigned stride0) {
  i32x8 g;
  const unsigned w0 = (1u << 16)               // data_size = 2 bytes
                      | (1u << 20)             // pad_enable
                      | (3u << 22);            // pad every 16 DWORDs, 1 DWORD
  g[0] = (int)w0;
  g[1] = (int)((tdim0 & 0xffffu) << 16);
  g[2] = (int)((tdim0 >> 16) | ((tdim1 & 0xffffu) << 16));
  g[3] = (int)((tdim1 >> 16) | (tile0 << 16));
  g[4] = (int)(tile1 & 0xffffu);               // tile_dim1 ; tile_dim2=0
  g[5] = (int)stride0;                         // dim0 stride (low 32)
  g[6] = 0;
  g[7] = 0;
  return g;
}

__device__ __forceinline__ void tdm_load(u32x4 g0, i32x8 g1) {
  const i32x4 z4 = {0, 0, 0, 0};
#if __clang_major__ >= 23
  const i32x8 z8 = {0, 0, 0, 0, 0, 0, 0, 0};
  __builtin_amdgcn_tensor_load_to_lds(g0, g1, z4, z4, z8, 0);
#else
  __builtin_amdgcn_tensor_load_to_lds(g0, g1, z4, z4, 0);
#endif
}

// ---------------------------------------------------------------------------
// Generic GEMM: C[M,N] = A[M,K] (f32, row-major) x B[N,K]^T (f32, row-major)
// Staging gathers all global vectors into registers first (loads clause up,
// one loadcnt wait covers them), then converts fp32->bf16 into padded LDS.
// ---------------------------------------------------------------------------
__global__ __launch_bounds__(256)
void gemm_xwT(const float* __restrict__ A, const float* __restrict__ B,
              float* __restrict__ C, int N, int K) {
  __shared__ __bf16 As[128][34];
  __shared__ __bf16 Bs[64][34];

  const int tid  = threadIdx.x;
  const int wave = tid >> 5;
  const int lane = tid & 31;
  const int half = lane >> 4;
  const int ln   = lane & 15;
  const int row0 = blockIdx.y * 128;
  const int col0 = blockIdx.x * 64;

  v8f acc[4];
#pragma unroll
  for (int nt = 0; nt < 4; ++nt)
#pragma unroll
    for (int e = 0; e < 8; ++e) acc[nt][e] = 0.0f;

  for (int k0 = 0; k0 < K; k0 += 32) {
    // ---- gather phase: 6 float4 per thread into registers ----
    float4 fa[4], fb[2];
#pragma unroll
    for (int i = 0; i < 4; ++i) {
      const int idx = tid + i * 256;
      fa[i] = *reinterpret_cast<const float4*>(
          A + (size_t)(row0 + (idx >> 3)) * K + k0 + ((idx & 7) << 2));
    }
#pragma unroll
    for (int i = 0; i < 2; ++i) {
      const int idx = tid + i * 256;
      fb[i] = *reinterpret_cast<const float4*>(
          B + (size_t)(col0 + (idx >> 3)) * K + k0 + ((idx & 7) << 2));
    }
    if (k0 + 32 < K) {  // L2 prefetch of next chunk (global_prefetch_b8)
      __builtin_prefetch(
          A + (size_t)(row0 + (tid >> 3)) * K + k0 + 32 + ((tid & 7) << 2), 0, 1);
      __builtin_prefetch(
          B + (size_t)(col0 + (tid >> 3)) * K + k0 + 32 + ((tid & 7) << 2), 0, 1);
    }
    // ---- convert + store phase ----
#pragma unroll
    for (int i = 0; i < 4; ++i) {
      const int idx = tid + i * 256;
      const int r   = idx >> 3;
      const int c4  = (idx & 7) << 2;
      As[r][c4 + 0] = (__bf16)fa[i].x;
      As[r][c4 + 1] = (__bf16)fa[i].y;
      As[r][c4 + 2] = (__bf16)fa[i].z;
      As[r][c4 + 3] = (__bf16)fa[i].w;
    }
#pragma unroll
    for (int i = 0; i < 2; ++i) {
      const int idx = tid + i * 256;
      const int r   = idx >> 3;
      const int c4  = (idx & 7) << 2;
      Bs[r][c4 + 0] = (__bf16)fb[i].x;
      Bs[r][c4 + 1] = (__bf16)fb[i].y;
      Bs[r][c4 + 2] = (__bf16)fb[i].z;
      Bs[r][c4 + 3] = (__bf16)fb[i].w;
    }
    __syncthreads();

    const v16bf a = frag_a_from_row(&As[wave * 16 + ln][0], half);
#pragma unroll
    for (int nt = 0; nt < 4; ++nt) {
      const v16bf b = frag_b_from_row(&Bs[nt * 16 + ln][0], half);
      acc[nt] = wmma_bf16(a, b, acc[nt]);
    }
    __syncthreads();
  }

#pragma unroll
  for (int nt = 0; nt < 4; ++nt)
#pragma unroll
    for (int r = 0; r < 8; ++r)
      C[(size_t)(row0 + wave * 16 + r + 8 * half) * N + col0 + nt * 16 + ln] =
          acc[nt][r];
}

// ---------------------------------------------------------------------------
// RMSNorm
// ---------------------------------------------------------------------------
__global__ __launch_bounds__(256)
void rms_kernel(const float* __restrict__ x, int ldx, const float* __restrict__ w,
                float* __restrict__ y, int ldy, int width) {
  const int row = blockIdx.x;
  const int tid = threadIdx.x;
  const float* xr = x + (size_t)row * ldx;

  float ss = 0.0f;
  for (int i = tid; i < width; i += 256) {
    const float v = xr[i];
    ss += v * v;
  }
  __shared__ float red[256];
  red[tid] = ss;
  __syncthreads();
#pragma unroll
  for (int s = 128; s > 0; s >>= 1) {
    if (tid < s) red[tid] += red[tid + s];
    __syncthreads();
  }
  const float inv = rsqrtf(red[0] / (float)width + 1e-6f);
  for (int i = tid; i < width; i += 256)
    y[(size_t)row * ldy + i] = w[i] * (xr[i] * inv);
}

// ---------------------------------------------------------------------------
// YaRN RoPE + pack: Qb[h][s][192], Kb[h][s][192], Vt[h][d][s] (bf16)
// ---------------------------------------------------------------------------
__global__ __launch_bounds__(256)
void rope_pack(const float* __restrict__ qf, const float* __restrict__ kvf,
               const float* __restrict__ ckv, const int* __restrict__ pos,
               __bf16* __restrict__ Qb, __bf16* __restrict__ Kb,
               __bf16* __restrict__ Vt) {
  const int s   = blockIdx.x;
  const int tid = threadIdx.x;

  __shared__ float cs[32], sn[32], kp[64];

  if (tid < 32) {
    const int i   = tid;
    const float p = (float)pos[s];
    const float f    = (float)i / 32.0f;
    const float fe   = powf(50000.0f, -f);
    const float fi   = fe / 32.0f;
    const float l2b  = 2.0f * logf(50000.0f);
    const float twoPi = 6.283185307179586f;
    float low  = floorf(64.0f * logf(4096.0f / (32.0f * twoPi)) / l2b);
    float high = ceilf(64.0f * logf(4096.0f / (1.0f * twoPi)) / l2b);
    low  = fmaxf(low, 0.0f);
    high = fminf(high, 63.0f);
    const float hieff = (low == high) ? (high + 0.001f) : high;
    const float ramp  = fminf(fmaxf(((float)i - low) / (hieff - low), 0.0f), 1.0f);
    const float inv_freq = fi * ramp + fe * (1.0f - ramp);
    const float ang = p * inv_freq;
    cs[i] = cosf(ang);
    sn[i] = sinf(ang);
    const float x0 = ckv[(size_t)s * CKVW_ + KVLR_ + 2 * i];
    const float x1 = ckv[(size_t)s * CKVW_ + KVLR_ + 2 * i + 1];
    kp[i]      = x0 * cs[i] - x1 * sn[i];
    kp[32 + i] = x1 * cs[i] + x0 * sn[i];
  }
  __syncthreads();

  for (int idx = tid; idx < NH_ * QD_; idx += 256) {
    const int h = idx / QD_;
    const int d = idx - h * QD_;
    const float* qrow = qf + (size_t)s * NQD_ + (size_t)h * QD_;
    float val;
    if (d < DN_) {
      val = qrow[d];
    } else {
      const int j = d - DN_;
      const int i = j & 31;
      const float x0 = qrow[DN_ + 2 * i];
      const float x1 = qrow[DN_ + 2 * i + 1];
      val = (j < 32) ? (x0 * cs[i] - x1 * sn[i]) : (x1 * cs[i] + x0 * sn[i]);
    }
    Qb[((size_t)h * S_LEN + s) * QD_ + d] = (__bf16)val;
  }
  for (int idx = tid; idx < NH_ * QD_; idx += 256) {
    const int h = idx / QD_;
    const int d = idx - h * QD_;
    const float val = (d < DN_)
                          ? kvf[(size_t)s * NKV_ + (size_t)h * 256 + d]
                          : kp[d - DN_];
    Kb[((size_t)h * S_LEN + s) * QD_ + d] = (__bf16)val;
  }
  for (int idx = tid; idx < NH_ * DV_; idx += 256) {
    const int h = idx / DV_;
    const int d = idx - h * DV_;
    Vt[((size_t)h * DV_ + d) * S_LEN + s] =
        (__bf16)kvf[(size_t)s * NKV_ + (size_t)h * 256 + DN_ + d];
  }
}

// ---------------------------------------------------------------------------
// Flash attention (causal), double-buffered TDM pipeline:
//   wave 0 issues K/V descriptors for tile kt+1 into the ping-pong buffer,
//   then waits TENSORcnt<=2 (only the 2 just-issued transfers outstanding),
//   guaranteeing tile kt has landed; all 8 waves compute tile kt while the
//   Tensor Data Mover fills tile kt+1.
// TDM padding (1 DWORD / 16 DWORDs) -> conflict-free row strides:
//   K tile row stride = 408 B (32 rows x 192 bf16)
//   V tile row stride =  68 B (128 rows x 32 bf16)
// ---------------------------------------------------------------------------
#define KS_ROW_BYTES 408  // 192*2 + (384/64)*4
#define VS_ROW_BYTES 68   // 32*2 + 4
#define KS_BYTES (32 * KS_ROW_BYTES + 8)
#define VS_BYTES (128 * VS_ROW_BYTES + 8)

__global__ __launch_bounds__(256)
void flash_attn(const __bf16* __restrict__ Qb, const __bf16* __restrict__ Kb,
                const __bf16* __restrict__ Vt, float* __restrict__ ao) {
  const int h    = blockIdx.x;
  const int qb0  = blockIdx.y * 128;
  const int tid  = threadIdx.x;
  const int wave = tid >> 5;
  const int lane = tid & 31;
  const int half = lane >> 4;
  const int ln   = lane & 15;
  const int q0   = qb0 + wave * 16;

  __shared__ __align__(16) char KsRaw[2][KS_BYTES];
  __shared__ __align__(16) char VsRaw[2][VS_BYTES];
  __shared__ __bf16 Ps[8][16][34];

  const float mscale = 0.1f * logf(32.0f) + 1.0f;
  const float scl    = mscale * mscale * rsqrtf((float)QD_);

  // Q fragments for this wave's 16 rows, kept in VGPRs across the key loop
  const __bf16* qrow = Qb + ((size_t)h * S_LEN + (size_t)(q0 + ln)) * QD_;
  v16bf qfr[6];
#pragma unroll
  for (int dc = 0; dc < 6; ++dc) qfr[dc] = frag_a_from_row(qrow + dc * 32, half);

  float run_max[8], run_sum[8];
  v8f accv[8];
#pragma unroll
  for (int r = 0; r < 8; ++r) {
    run_max[r] = -3.0e38f;
    run_sum[r] = 0.0f;
  }
#pragma unroll
  for (int nt = 0; nt < 8; ++nt)
#pragma unroll
    for (int e = 0; e < 8; ++e) accv[nt][e] = 0.0f;

  const unsigned ldsK[2] = {LDS_OFF(KsRaw[0]), LDS_OFF(KsRaw[1])};
  const unsigned ldsV[2] = {LDS_OFF(VsRaw[0]), LDS_OFF(VsRaw[1])};

  const int ktiles_blk = ((qb0 + 127) >> 5) + 1;  // block-uniform trip count
  const int my_ktiles  = ((q0 + 15) >> 5) + 1;    // this wave's causal limit

  // Issue DMA for K/V tiles of a 32-key chunk into buffer `buf`
  auto issue_tile = [&](int kb, int buf) {
    tdm_load(tdm_g0(ldsK[buf], Kb + ((size_t)h * S_LEN + kb) * QD_),
             tdm_g1_2d(/*tdim0=*/QD_, /*tdim1=*/S_LEN,
                       /*tile0=*/QD_, /*tile1=*/32, /*stride0=*/QD_));
    tdm_load(tdm_g0(ldsV[buf], Vt + (size_t)h * DV_ * S_LEN + kb),
             tdm_g1_2d(/*tdim0=*/S_LEN, /*tdim1=*/DV_,
                       /*tile0=*/32, /*tile1=*/DV_, /*stride0=*/S_LEN));
  };

  if (wave == 0) issue_tile(0, 0);  // prologue: tile 0 in flight

  for (int kt = 0; kt < ktiles_blk; ++kt) {
    const int kb  = kt << 5;
    const int buf = kt & 1;

    if (wave == 0) {
      if (kt + 1 < ktiles_blk) {
        issue_tile((kt + 1) << 5, buf ^ 1);       // prefetch next tile
        __builtin_amdgcn_s_wait_tensorcnt(2);     // tile kt landed
      } else {
        __builtin_amdgcn_s_wait_tensorcnt(0);     // drain
      }
    }
    __syncthreads();  // tile kt visible to all waves

    if (kt < my_ktiles) {
      const char* ks = KsRaw[buf];
      const char* vs = VsRaw[buf];

      // ---- scores: S[16q x 32k] ----
      v8f s0, s1;
#pragma unroll
      for (int e = 0; e < 8; ++e) {
        s0[e] = 0.0f;
        s1[e] = 0.0f;
      }
#pragma unroll
      for (int dc = 0; dc < 6; ++dc) {
        // element (row,d): byte = row*408 + 2d + 4*(d>>5); d=dc*32 starts a
        // 32-elem group, so the 32-elem window is contiguous
        const __bf16* kr0 =
            (const __bf16*)(ks + ln * KS_ROW_BYTES + dc * 64 + (dc * 2) * 4);
        const __bf16* kr1 = (const __bf16*)(ks + (16 + ln) * KS_ROW_BYTES +
                                            dc * 64 + (dc * 2) * 4);
        const v16bf b0 = frag_b_from_row(kr0, half);
        const v16bf b1 = frag_b_from_row(kr1, half);
        s0 = wmma_bf16(qfr[dc], b0, s0);
        s1 = wmma_bf16(qfr[dc], b1, s1);
      }

      // ---- online softmax ----
      float corr[8];
#pragma unroll
      for (int r = 0; r < 8; ++r) {
        const int q = q0 + r + 8 * half;
        float v0 = s0[r] * scl + (((kb + ln) <= q) ? 0.0f : -1.0e9f);
        float v1 = s1[r] * scl + (((kb + 16 + ln) <= q) ? 0.0f : -1.0e9f);
        float m = fmaxf(v0, v1);
#pragma unroll
        for (int off = 1; off < 16; off <<= 1)
          m = fmaxf(m, __shfl_xor(m, off, 32));
        const float nm = fmaxf(run_max[r], m);
        const float p0 = __expf(v0 - nm);
        const float p1 = __expf(v1 - nm);
        float ps = p0 + p1;
#pragma unroll
        for (int off = 1; off < 16; off <<= 1) ps += __shfl_xor(ps, off, 32);
        corr[r]    = __expf(run_max[r] - nm);
        run_sum[r] = run_sum[r] * corr[r] + ps;
        run_max[r] = nm;
        Ps[wave][r + 8 * half][ln]      = (__bf16)p0;
        Ps[wave][r + 8 * half][16 + ln] = (__bf16)p1;
      }
#pragma unroll
      for (int nt = 0; nt < 8; ++nt)
#pragma unroll
        for (int e = 0; e < 8; ++e) accv[nt][e] *= corr[e];

      // wave-local LDS fence, then re-read P in the A-fragment layout
      asm volatile("s_wait_dscnt 0x0" ::: "memory");
      const v16bf aP = frag_a_from_row(&Ps[wave][ln][0], half);

      // ---- PV: accumulate 16x128 ----
#pragma unroll
      for (int nt = 0; nt < 8; ++nt) {
        const __bf16* vr = (const __bf16*)(vs + (nt * 16 + ln) * VS_ROW_BYTES);
        const v16bf bV = frag_b_from_row(vr, half);
        accv[nt] = wmma_bf16(aP, bV, accv[nt]);
      }
      asm volatile("s_wait_dscnt 0x0" ::: "memory");
    }
    __syncthreads();  // everyone done with buf before it is refilled
  }

  // ---- epilogue ----
#pragma unroll
  for (int r = 0; r < 8; ++r) {
    const int q    = q0 + r + 8 * half;
    const float iv = 1.0f / run_sum[r];
#pragma unroll
    for (int nt = 0; nt < 8; ++nt)
      ao[(size_t)q * (NH_ * DV_) + (size_t)h * DV_ + nt * 16 + ln] =
          accv[nt][r] * iv;
  }
}

// ---------------------------------------------------------------------------
// kernel_launch
// ---------------------------------------------------------------------------
extern "C" void kernel_launch(void* const* d_in, const int* in_sizes, int n_in,
                              void* d_out, int out_size, void* d_ws,
                              size_t ws_size, hipStream_t stream) {
  (void)in_sizes;
  (void)n_in;
  (void)out_size;
  (void)ws_size;

  const float* hs      = (const float*)d_in[0];
  const int*   pos     = (const int*)d_in[2];
  const float* q_a_w   = (const float*)d_in[3];
  const float* q_a_ln  = (const float*)d_in[4];
  const float* q_b_w   = (const float*)d_in[5];
  const float* kv_a_w  = (const float*)d_in[6];
  const float* kv_a_ln = (const float*)d_in[7];
  const float* kv_b_w  = (const float*)d_in[8];
  const float* o_w     = (const float*)d_in[9];
  float*       out     = (float*)d_out;

  char*  ws  = (char*)d_ws;
  size_t off = 0;
  auto take = [&](size_t bytes) -> char* {
    char* p = ws + off;
    off += (bytes + 255) & ~(size_t)255;
    return p;
  };

  float*  qa     = (float*)take((size_t)S_LEN * QLR_ * 4);
  float*  qa_n   = (float*)take((size_t)S_LEN * QLR_ * 4);
  float*  qfull  = (float*)take((size_t)S_LEN * NQD_ * 4);
  float*  ckv    = (float*)take((size_t)S_LEN * CKVW_ * 4);
  float*  comp_n = (float*)take((size_t)S_LEN * KVLR_ * 4);
  float*  kvfull = (float*)take((size_t)S_LEN * NKV_ * 4);
  __bf16* Qb     = (__bf16*)take((size_t)NH_ * S_LEN * QD_ * 2);
  __bf16* Kb     = (__bf16*)take((size_t)NH_ * S_LEN * QD_ * 2);
  __bf16* Vt     = (__bf16*)take((size_t)NH_ * DV_ * S_LEN * 2);
  float*  ao     = (float*)take((size_t)S_LEN * NH_ * DV_ * 4);

  const dim3 blk(256);

  gemm_xwT<<<dim3(QLR_ / 64, S_LEN / 128), blk, 0, stream>>>(hs, q_a_w, qa,
                                                             QLR_, H_DIM);
  gemm_xwT<<<dim3(CKVW_ / 64, S_LEN / 128), blk, 0, stream>>>(hs, kv_a_w, ckv,
                                                              CKVW_, H_DIM);
  rms_kernel<<<S_LEN, blk, 0, stream>>>(qa, QLR_, q_a_ln, qa_n, QLR_, QLR_);
  rms_kernel<<<S_LEN, blk, 0, stream>>>(ckv, CKVW_, kv_a_ln, comp_n, KVLR_,
                                        KVLR_);
  gemm_xwT<<<dim3(NQD_ / 64, S_LEN / 128), blk, 0, stream>>>(qa_n, q_b_w, qfull,
                                                             NQD_, QLR_);
  gemm_xwT<<<dim3(NKV_ / 64, S_LEN / 128), blk, 0, stream>>>(comp_n, kv_b_w,
                                                             kvfull, NKV_, KVLR_);
  rope_pack<<<S_LEN, blk, 0, stream>>>(qfull, kvfull, ckv, pos, Qb, Kb, Vt);
  flash_attn<<<dim3(NH_, S_LEN / 128), blk, 0, stream>>>(Qb, Kb, Vt, ao);
  gemm_xwT<<<dim3(H_DIM / 64, S_LEN / 128), blk, 0, stream>>>(ao, o_w, out,
                                                              H_DIM, NH_ * DV_);
}
